// BiLSTMTagger_4123168604853
// MI455X (gfx1250) — matrix operand order repaired and checked by reading, not compiled, not run on hardware
//
#include <hip/hip_runtime.h>
#include <hip/hip_bf16.h>

// ---------------------------------------------------------------------------
// BiLSTM tagger on gfx1250 (MI455X), bf16 WMMA path.
//   - x-projections precomputed for all timesteps (parallel GEMM)
//   - persistent recurrent scan, fwd+bwd concurrently
//   - per-step h strip staged into LDS via the Tensor Data Mover (TDM)
//   B=64, S=256, E=H=1024, T=50.
// ---------------------------------------------------------------------------

typedef __attribute__((ext_vector_type(16))) __bf16   v16bf;
typedef __attribute__((ext_vector_type(8)))  float    v8f;
typedef __attribute__((ext_vector_type(4)))  unsigned v4u;
typedef __attribute__((ext_vector_type(8)))  int      v8i;
typedef __attribute__((ext_vector_type(4)))  int      v4i;

static constexpr int  Bb  = 64;
static constexpr int  Ss  = 256;
static constexpr int  Ee  = 1024;
static constexpr int  Hh  = 1024;
static constexpr int  Tt  = 50;
static constexpr long BS  = (long)Bb * Ss;     // 16384 rows
static constexpr int  G4H = 4 * Hh;            // 4096 gate columns

// Persistent-scan shape: per direction 16 blocks x 512 threads = 256 waves.
static constexpr int  NBLK = 16;
static constexpr int  NTHR = 512;
static constexpr int  HALF_THREADS = NBLK * NTHR;   // 8192

// LDS h-strip written by TDM with padding: row = 512 dwords data + 2x2 dword
// pads (one mid-row, one end-of-row) -> pitch 516 dwords = 1032 elements.
// Rows land 4 LDS banks apart -> conflict-free b128 fragment reads.
static constexpr int  HPITCH = 1032;

// ---------------------------------------------------------------------------
// WMMA fragment loaders (layouts per CDNA5 ISA 7.12.2, wave32)
// A: 16x32 bf16.  lane l: M = l&15, Khalf = l>>4.
//   regs 0..3 = K (khalf*8 + 0..7), regs 4..7 = K (16 + khalf*8 + 0..7)
// B: 32x16 bf16, stored TRANSPOSED as Wt[N][K].
//   lane l: N = l&15, Khalf = l>>4; regs 0..3 = K (khalf*8 + 0..7), regs 4..7 = K (16 + khalf*8 + 0..7).
// C/D: f32, reg r -> M = r + 8*(l>>4), N = l&15.
// ---------------------------------------------------------------------------

__device__ __forceinline__ v16bf load_a_frag(const __bf16* __restrict__ a00,
                                             long row_stride, int lane) {
  const __bf16* p = a00 + (long)(lane & 15) * row_stride + ((lane >> 4) << 3);
  v16bf a;
  ((uint4*)&a)[0] = *(const uint4*)(p);
  ((uint4*)&a)[1] = *(const uint4*)(p + 16);
  return a;
}

__device__ __forceinline__ v16bf load_a_frag_lds(const __bf16* p0, int lane) {
  // p0 points at (row 0, element kk) inside the padded LDS strip
  const __bf16* p = p0 + (lane & 15) * HPITCH + ((lane >> 4) << 3);
  v16bf a;
  ((uint4*)&a)[0] = *(const uint4*)(p);
  ((uint4*)&a)[1] = *(const uint4*)(p + 16);
  return a;
}

__device__ __forceinline__ v16bf load_b_frag(const __bf16* __restrict__ b00,
                                             long kdim, int lane) {
  const __bf16* p = b00 + (long)(lane & 15) * kdim + ((lane >> 4) << 3);
  v16bf b;
  ((uint4*)&b)[0] = *(const uint4*)(p);
  ((uint4*)&b)[1] = *(const uint4*)(p + 16);
  return b;
}

// ---------------------------------------------------------------------------
// Software grid barrier over one 16-block half (monotone counter, agent scope).
// ---------------------------------------------------------------------------
__device__ __forceinline__ void grid_barrier(unsigned* __restrict__ bar,
                                             unsigned goal) {
  __threadfence();
  __syncthreads();
  if (threadIdx.x == 0) {
    __hip_atomic_fetch_add(bar, 1u, __ATOMIC_ACQ_REL, __HIP_MEMORY_SCOPE_AGENT);
    while (__hip_atomic_load(bar, __ATOMIC_ACQUIRE, __HIP_MEMORY_SCOPE_AGENT) < goal) {
      __builtin_amdgcn_s_sleep(1);
    }
  }
  __syncthreads();
  __threadfence();
}

// ---------------------------------------------------------------------------
// Setup kernels
// ---------------------------------------------------------------------------

__global__ void embed_cast_kernel(const int* __restrict__ words,
                                  const float* __restrict__ emb,
                                  __bf16* __restrict__ X) {
  long idx = (long)blockIdx.x * blockDim.x + threadIdx.x;   // BS * E threads
  if (idx >= BS * (long)Ee) return;
  long row = idx >> 10;
  int  e   = (int)(idx & 1023);
  int  w   = words[row];
  X[idx] = (__bf16)emb[(long)w * Ee + e];
}

// W: f32 [4][D+H][H]  ->  WxT bf16 [4H][D], WhT bf16 [4H][H]
__global__ void cast_lstm_w_kernel(const float* __restrict__ W, int D,
                                   __bf16* __restrict__ WxT,
                                   __bf16* __restrict__ WhT) {
  long idx = (long)blockIdx.x * blockDim.x + threadIdx.x;
  long per_gate = (long)(D + Hh) * Hh;
  if (idx >= 4 * per_gate) return;
  int  g   = (int)(idx / per_gate);
  long rem = idx - (long)g * per_gate;
  int  k   = (int)(rem / Hh);
  int  h   = (int)(rem - (long)(rem / Hh) * Hh);
  float v  = W[idx];
  long  n  = (long)g * Hh + h;
  if (k < D) WxT[n * D + k]        = (__bf16)v;
  else       WhT[n * Hh + (k - D)] = (__bf16)v;
}

// Wout: f32 [2H][T] -> WoutT bf16 [64][2H]  (rows >= T zero-padded)
__global__ void cast_wout_kernel(const float* __restrict__ Wout,
                                 __bf16* __restrict__ WoutT) {
  long idx = (long)blockIdx.x * blockDim.x + threadIdx.x;   // 64 * 2048
  if (idx >= 64L * 2048) return;
  int n = (int)(idx / 2048);
  int k = (int)(idx - (long)n * 2048);
  WoutT[idx] = (n < Tt) ? (__bf16)Wout[(long)k * Tt + n] : (__bf16)0.0f;
}

__global__ void zero_bar_kernel(unsigned* __restrict__ bar) {
  if (threadIdx.x < 8) bar[threadIdx.x] = 0u;
}

// ---------------------------------------------------------------------------
// x-projection (parallel over all B*S rows), stored row-major bf16:
//   Pre[b*S+s][n] = bias[n] + A[b*S+s, :] @ WxT[n, :]
// Each wave: 64-row strip x one 16-col N-tile (B fragment reused 4x).
// Grid: (BS/64) * (4096/16) = 65536 waves -> 8192 blocks x 256 threads.
// ---------------------------------------------------------------------------
__global__ void xproj_kernel(const __bf16* __restrict__ A, long Din,
                             const __bf16* __restrict__ WxT,
                             const float*  __restrict__ bias,
                             __bf16* __restrict__ Pre) {
  int w    = (blockIdx.x * blockDim.x + threadIdx.x) >> 5;  // 0..65535
  int lane = threadIdx.x & 31;
  int mstrip = w >> 8;             // 0..255 -> rows mstrip*64 ..+63
  int n0     = (w & 255) << 4;
  int n      = n0 + (lane & 15);

  float bv = bias[n];
  v8f acc[4];
#pragma unroll
  for (int m = 0; m < 4; ++m)
#pragma unroll
    for (int r = 0; r < 8; ++r) acc[m][r] = bv;

  const __bf16* a00 = A + (long)mstrip * 64 * Din;
  const __bf16* b00 = WxT + (long)n0 * Din;
#pragma unroll 2
  for (long k0 = 0; k0 < Din; k0 += 32) {
    // hoist all loads so WMMAs can overlap outstanding loads
    v16bf bf = load_b_frag(b00 + k0, Din, lane);
    v16bf a0 = load_a_frag(a00 +  0 * Din + k0, Din, lane);
    v16bf a1 = load_a_frag(a00 + 16 * Din + k0, Din, lane);
    v16bf a2 = load_a_frag(a00 + 32 * Din + k0, Din, lane);
    v16bf a3 = load_a_frag(a00 + 48 * Din + k0, Din, lane);
    acc[0] = __builtin_amdgcn_wmma_f32_16x16x32_bf16(false, a0, false, bf,
                                                     (short)0, acc[0], false, false);
    acc[1] = __builtin_amdgcn_wmma_f32_16x16x32_bf16(false, a1, false, bf,
                                                     (short)0, acc[1], false, false);
    acc[2] = __builtin_amdgcn_wmma_f32_16x16x32_bf16(false, a2, false, bf,
                                                     (short)0, acc[2], false, false);
    acc[3] = __builtin_amdgcn_wmma_f32_16x16x32_bf16(false, a3, false, bf,
                                                     (short)0, acc[3], false, false);
  }

  int off8 = (lane >> 4) << 3;
#pragma unroll
  for (int m = 0; m < 4; ++m) {
#pragma unroll
    for (int r = 0; r < 8; ++r) {
      long row = (long)mstrip * 64 + m * 16 + off8 + r;
      Pre[row * G4H + n] = (__bf16)acc[m][r];   // lanes 0..15 contiguous in n
    }
  }
}

// ---------------------------------------------------------------------------
// Persistent recurrent scan, BOTH directions concurrently.
//   blocks 0..15  : forward  (half 0)
//   blocks 16..31 : backward (half 1)
// Per step (per half): TDM-load h (64x1024 bf16, padded) into LDS, seed acc
// from precomputed x-projection, accumulate h@WhT with WMMA (K=1024),
// grid-barrier, gates, grid-barrier. Barriers are per-half.
// ---------------------------------------------------------------------------
__global__ __launch_bounds__(NTHR, 1)
void lstm_layer_kernel(const __bf16* __restrict__ preF,
                       const __bf16* __restrict__ preB,
                       const __bf16* __restrict__ WhF,
                       const __bf16* __restrict__ WhB,
                       float* __restrict__ cstF, float* __restrict__ cstB,
                       __bf16* __restrict__ hstF, __bf16* __restrict__ hstB,
                       float* __restrict__ GF, float* __restrict__ GB,
                       __bf16* __restrict__ layer_out,
                       unsigned* __restrict__ bar2) {
  __shared__ __bf16 hlds[64 * HPITCH];            // 129 KB of the 320 KB LDS

  const int dir = blockIdx.x >> 4;                // 0 = fwd, 1 = bwd
  const int blk = blockIdx.x & 15;
  const int tid = threadIdx.x;
  const int htid = blk * NTHR + tid;              // 0..8191 within half
  const int lane = tid & 31;
  const int w    = (blk << 4) + (tid >> 5);       // wave id in half: 0..255
  const int n0   = w << 4;
  const int n    = n0 + (lane & 15);
  const int off8 = (lane >> 4) << 3;

  const __bf16* pre = dir ? preB : preF;
  const __bf16* Wh  = dir ? WhB  : WhF;
  float*        cst = dir ? cstB : cstF;
  __bf16*       hst = dir ? hstB : hstF;
  float*        G   = dir ? GB   : GF;
  unsigned*     bar = bar2 + dir;

  unsigned goal = 0;

  // ---- zero c / h state for this half; visibly initialize the LDS strip so
  //      the compiler cannot treat it as never-stored (TDM writes are opaque)
  for (int idx = htid; idx < Bb * Hh; idx += HALF_THREADS) {
    cst[idx] = 0.0f;
    hst[idx] = (__bf16)0.0f;
  }
  for (int i = tid; i < 64 * HPITCH; i += NTHR)
    hlds[i] = (__bf16)0.0f;
  __syncthreads();
  goal += NBLK; grid_barrier(bar, goal);

  // ---- TDM descriptor: 2D tensor 512 dwords x 64 rows, 4B elements,
  //      pad 2 dwords every 256 dwords -> LDS pitch 516 dwords ----
  unsigned long long gaddr = (unsigned long long)hst;
  v4u g0;
  g0[0] = 1u;                                   // count=1, normal load D#
  g0[1] = 0u;                                   // lds_addr = start of hlds
  g0[2] = (unsigned)(gaddr & 0xFFFFFFFFu);      // global_addr[31:0]
  g0[3] = (unsigned)((gaddr >> 32) & 0x01FFFFFFu) | (2u << 30);  // [56:32],type=2
  v8i g1;
  g1[0] = (2 << 16)        // data_size = 4B
        | (1 << 20)        // pad_enable
        | (7 << 22)        // pad_interval: 256 dwords
        | (1 << 25);       // pad_amount: 2 dwords
  g1[1] = (int)(512u << 16);   // tensor_dim0 = 512 (low16 at bit48)
  g1[2] = (int)(64u  << 16);   // tensor_dim1 = 64  (low16 at bit80)
  g1[3] = (int)(512u << 16);   // tile_dim0 = 512   (bits 127:112)
  g1[4] = 64;                  // tile_dim1 = 64
  g1[5] = 512;                 // tensor_dim0_stride = 512 dwords
  g1[6] = 0;
  g1[7] = 0;
  v4i z4; z4[0] = 0; z4[1] = 0; z4[2] = 0; z4[3] = 0;
  v8i z8; z8[0] = 0; z8[1] = 0; z8[2] = 0; z8[3] = 0;
  z8[4] = 0; z8[5] = 0; z8[6] = 0; z8[7] = 0;

  const __bf16* bH = Wh + (long)n0 * Hh;

  for (int t = 0; t < Ss; ++t) {
    const int s = dir ? (Ss - 1 - t) : t;

    // ---- stage h strip into LDS via Tensor Data Mover ----
    if ((tid >> 5) == 0) {
      __builtin_amdgcn_tensor_load_to_lds(g0, g1, z4, z4, z8, 0);
      __builtin_amdgcn_s_wait_tensorcnt(0);
    }
    __syncthreads();
    // opaque barrier: TDM wrote hlds; forbid folding/forwarding of LDS reads
    asm volatile("" ::: "memory");

    // ---- seed accumulators from precomputed x-projection (row-major) ----
    const __bf16* Ps = pre + (long)s * G4H + n;   // + row*(Ss*G4H)
    v8f acc[4];
#pragma unroll
    for (int m = 0; m < 4; ++m)
#pragma unroll
      for (int r = 0; r < 8; ++r)
        acc[m][r] = (float)Ps[(long)(m * 16 + off8 + r) * Ss * G4H];

    // ---- h contribution: K = 1024, A from LDS (TDM layout), B from L2 ----
#pragma unroll 2
    for (int k0 = 0; k0 < Hh; k0 += 32) {
      int kk = k0 + ((k0 >> 9) << 2);   // +4 elements past the mid-row pad
      v16bf bf = load_b_frag(bH + k0, (long)Hh, lane);
      v16bf a0 = load_a_frag_lds(&hlds[ 0 * HPITCH + kk], lane);
      v16bf a1 = load_a_frag_lds(&hlds[16 * HPITCH + kk], lane);
      v16bf a2 = load_a_frag_lds(&hlds[32 * HPITCH + kk], lane);
      v16bf a3 = load_a_frag_lds(&hlds[48 * HPITCH + kk], lane);
      acc[0] = __builtin_amdgcn_wmma_f32_16x16x32_bf16(false, a0, false, bf,
                                                       (short)0, acc[0], false, false);
      acc[1] = __builtin_amdgcn_wmma_f32_16x16x32_bf16(false, a1, false, bf,
                                                       (short)0, acc[1], false, false);
      acc[2] = __builtin_amdgcn_wmma_f32_16x16x32_bf16(false, a2, false, bf,
                                                       (short)0, acc[2], false, false);
      acc[3] = __builtin_amdgcn_wmma_f32_16x16x32_bf16(false, a3, false, bf,
                                                       (short)0, acc[3], false, false);
    }

    // ---- store gate pre-activations ----
#pragma unroll
    for (int m = 0; m < 4; ++m) {
      int mrow = m * 16 + off8;
#pragma unroll
      for (int r = 0; r < 8; ++r)
        G[(long)(mrow + r) * G4H + n] = acc[m][r];
    }
    goal += NBLK; grid_barrier(bar, goal);

    // ---- pointwise gates + state update ----
    for (int idx = htid; idx < Bb * Hh; idx += HALF_THREADS) {
      int b  = idx >> 10;
      int hh = idx & 1023;
      const float* g = G + (long)b * G4H;
      float f  = 1.0f / (1.0f + __expf(-g[hh]));
      float i  = 1.0f / (1.0f + __expf(-g[Hh + hh]));
      float ch = tanhf(g[2 * Hh + hh]);
      float o  = 1.0f / (1.0f + __expf(-g[3 * Hh + hh]));
      float cn = f * cst[idx] + i * ch;
      cst[idx] = cn;
      float hn = o * tanhf(cn);
      hst[idx] = (__bf16)hn;
      layer_out[((long)b * Ss + s) * 2048 + (long)dir * Hh + hh] = (__bf16)hn;
    }
    goal += NBLK; grid_barrier(bar, goal);
  }
}

// ---------------------------------------------------------------------------
// Output projection: logits[16384][50] = l2out[16384][2048] @ Wout + bout
// ---------------------------------------------------------------------------
__global__ void out_proj_kernel(const __bf16* __restrict__ l2out,
                                const __bf16* __restrict__ WoutT,
                                const float*  __restrict__ bout,
                                float* __restrict__ out) {
  int w    = (blockIdx.x * blockDim.x + threadIdx.x) >> 5;  // 0..4095
  int lane = threadIdx.x & 31;
  int m0   = (w >> 2) << 4;
  int n0   = (w & 3) << 4;

  v8f acc, acc2;
#pragma unroll
  for (int r = 0; r < 8; ++r) { acc[r] = 0.0f; acc2[r] = 0.0f; }

  const __bf16* a00 = l2out + (long)m0 * 2048;
  const __bf16* b00 = WoutT + (long)n0 * 2048;
  for (long k0 = 0; k0 < 2048; k0 += 64) {
    v16bf a0 = load_a_frag(a00 + k0,      2048L, lane);
    v16bf b0 = load_b_frag(b00 + k0,      2048L, lane);
    v16bf a1 = load_a_frag(a00 + k0 + 32, 2048L, lane);
    v16bf b1 = load_b_frag(b00 + k0 + 32, 2048L, lane);
    acc  = __builtin_amdgcn_wmma_f32_16x16x32_bf16(false, a0, false, b0,
                                                   (short)0, acc,  false, false);
    acc2 = __builtin_amdgcn_wmma_f32_16x16x32_bf16(false, a1, false, b1,
                                                   (short)0, acc2, false, false);
  }
#pragma unroll
  for (int r = 0; r < 8; ++r) acc[r] += acc2[r];

  int n = n0 + (lane & 15);
  if (n < Tt) {
    float bb = bout[n];
    int mrow = m0 + ((lane >> 4) << 3);
#pragma unroll
    for (int r = 0; r < 8; ++r)
      out[(long)(mrow + r) * Tt + n] = acc[r] + bb;
  }
}

// ---------------------------------------------------------------------------
// Host orchestration
// ---------------------------------------------------------------------------
extern "C" void kernel_launch(void* const* d_in, const int* in_sizes, int n_in,
                              void* d_out, int out_size, void* d_ws, size_t ws_size,
                              hipStream_t stream) {
  const int*   words = (const int*)  d_in[0];
  const float* emb   = (const float*)d_in[1];
  const float* W1f   = (const float*)d_in[2];
  const float* b1f   = (const float*)d_in[3];
  const float* W1b   = (const float*)d_in[4];
  const float* b1b   = (const float*)d_in[5];
  const float* W2f   = (const float*)d_in[6];
  const float* b2f   = (const float*)d_in[7];
  const float* W2b   = (const float*)d_in[8];
  const float* b2b   = (const float*)d_in[9];
  const float* Wout  = (const float*)d_in[10];
  const float* bout  = (const float*)d_in[11];
  float* out = (float*)d_out;

  // ---- carve workspace ----
  size_t off = 0;
  auto carve = [&](size_t bytes) -> void* {
    void* p = (char*)d_ws + off;
    off = (off + bytes + 255) & ~(size_t)255;
    return p;
  };
  __bf16*   Xbf   = (__bf16*)carve(BS * (size_t)Ee * 2);        // 32 MB
  __bf16*   Wx1f  = (__bf16*)carve((size_t)G4H * Ee * 2);
  __bf16*   Wh1f  = (__bf16*)carve((size_t)G4H * Hh * 2);
  __bf16*   Wx1b  = (__bf16*)carve((size_t)G4H * Ee * 2);
  __bf16*   Wh1b  = (__bf16*)carve((size_t)G4H * Hh * 2);
  __bf16*   Wx2f  = (__bf16*)carve((size_t)G4H * 2048 * 2);
  __bf16*   Wh2f  = (__bf16*)carve((size_t)G4H * Hh * 2);
  __bf16*   Wx2b  = (__bf16*)carve((size_t)G4H * 2048 * 2);
  __bf16*   Wh2b  = (__bf16*)carve((size_t)G4H * Hh * 2);
  __bf16*   WoutT = (__bf16*)carve((size_t)64 * 2048 * 2);
  __bf16*   l1out = (__bf16*)carve(BS * (size_t)2048 * 2);      // 64 MB
  __bf16*   l2out = (__bf16*)carve(BS * (size_t)2048 * 2);      // 64 MB
  __bf16*   pre0  = (__bf16*)carve(BS * (size_t)G4H * 2);       // 128 MB
  __bf16*   pre1  = (__bf16*)carve(BS * (size_t)G4H * 2);       // 128 MB
  float*    G0    = (float*) carve((size_t)Bb * G4H * 4);
  float*    G1    = (float*) carve((size_t)Bb * G4H * 4);
  float*    cst0  = (float*) carve((size_t)Bb * Hh * 4);
  float*    cst1  = (float*) carve((size_t)Bb * Hh * 4);
  __bf16*   hst0  = (__bf16*)carve((size_t)Bb * Hh * 2);
  __bf16*   hst1  = (__bf16*)carve((size_t)Bb * Hh * 2);
  unsigned* bar   = (unsigned*)carve(256);                      // 4 counters
  (void)ws_size; (void)n_in; (void)in_sizes; (void)out_size;

  // ---- setup ----
  zero_bar_kernel<<<1, 64, 0, stream>>>(bar);
  {
    long n = BS * (long)Ee;
    embed_cast_kernel<<<(int)((n + 255) / 256), 256, 0, stream>>>(words, emb, Xbf);
  }
  {
    long n1 = 4L * (Ee + Hh) * Hh;
    long n2 = 4L * (2048 + Hh) * Hh;
    cast_lstm_w_kernel<<<(int)((n1 + 255) / 256), 256, 0, stream>>>(W1f, Ee,   Wx1f, Wh1f);
    cast_lstm_w_kernel<<<(int)((n1 + 255) / 256), 256, 0, stream>>>(W1b, Ee,   Wx1b, Wh1b);
    cast_lstm_w_kernel<<<(int)((n2 + 255) / 256), 256, 0, stream>>>(W2f, 2048, Wx2f, Wh2f);
    cast_lstm_w_kernel<<<(int)((n2 + 255) / 256), 256, 0, stream>>>(W2b, 2048, Wx2b, Wh2b);
    cast_wout_kernel<<<512, 256, 0, stream>>>(Wout, WoutT);
  }

  // ---- layer 1: x-projections (parallel), then fwd+bwd scan (persistent) ----
  xproj_kernel<<<8192, 256, 0, stream>>>(Xbf, (long)Ee, Wx1f, b1f, pre0);
  xproj_kernel<<<8192, 256, 0, stream>>>(Xbf, (long)Ee, Wx1b, b1b, pre1);
  lstm_layer_kernel<<<2 * NBLK, NTHR, 0, stream>>>(pre0, pre1, Wh1f, Wh1b,
                                                   cst0, cst1, hst0, hst1,
                                                   G0, G1, l1out, bar + 0);

  // ---- layer 2 ----
  xproj_kernel<<<8192, 256, 0, stream>>>(l1out, 2048L, Wx2f, b2f, pre0);
  xproj_kernel<<<8192, 256, 0, stream>>>(l1out, 2048L, Wx2b, b2b, pre1);
  lstm_layer_kernel<<<2 * NBLK, NTHR, 0, stream>>>(pre0, pre1, Wh2f, Wh2b,
                                                   cst0, cst1, hst0, hst1,
                                                   G0, G1, l2out, bar + 2);

  // ---- output projection ----
  out_proj_kernel<<<512, 256, 0, stream>>>(l2out, WoutT, bout, out);
}
